// GRU_46334107189669
// MI455X (gfx1250) — matrix operand order repaired and checked by reading, not compiled, not run on hardware
//
#include <hip/hip_runtime.h>
#include <hip/hip_bf16.h>
#include <math.h>

// GRU: B=4096, IN=1024, HID=2048, K = IN+HID = 3072 for all three GEMMs.
// bf16 WMMA (16x16x32, f32 accum). A/B operands pre-swizzled into the exact
// CDNA5 WMMA per-lane fragment layout; GEMM stages 128x32 slabs into LDS with
// async global->LDS copies (double buffered), waves read via ds_load_b128.

typedef __attribute__((ext_vector_type(16))) __bf16 v16bf;
typedef __attribute__((ext_vector_type(8)))  float  v8f;
typedef __attribute__((__vector_size__(4 * sizeof(int)))) int v4i;

#define B_SZ   4096
#define IN_SZ  1024
#define HID_SZ 2048
#define K_SZ   3072
#define KT     96        // K / 32
#define MT_A   256       // B_SZ / 16
#define NT_W   128       // HID / 16
#define FRAG   512       // elements in one 16x32 bf16 fragment (32 lanes x 16)

#define W3S_ELEMS (3ull * NT_W * KT * FRAG)          // 3 swizzled weight mats
#define XHS_ELEMS ((unsigned long long)MT_A * KT * FRAG)
// rhx swizzled buffer follows xhs, same size as xhs

#if defined(__gfx1250__) &&                                         \
    __has_builtin(__builtin_amdgcn_global_load_async_to_lds_b128) && \
    __has_builtin(__builtin_amdgcn_s_wait_asynccnt)
#define USE_ASYNC_LDS 1
#else
#define USE_ASYNC_LDS 0
#endif

__device__ static inline void sched_fence() {
#if defined(__gfx1250__) && __has_builtin(__builtin_amdgcn_sched_barrier)
  __builtin_amdgcn_sched_barrier(0);
#endif
}

// --- ISA 7.12.2 fragment maps (lane l, halfword t -> K index) ---
__device__ static inline int amapK(int l, int t) {
  return t + 8 * ((t >= 8) + (l >= 16));
}
__device__ static inline int bmapK(int l, int t) {
  return t + 8 * ((t >= 8) + (l >= 16));
}

// ---------------- prep: convert + swizzle everything once per call ----------
// frag regions: [0, 36864): weights   (3 * 128 * 96)
//               [36864, 61440): xh    (256 * 96)
//               [61440, 69632): rhx x-part (256 * 32, kt in [64,96))
__global__ __launch_bounds__(256) void gru_prep(
    const float* __restrict__ x, const float* __restrict__ h,
    const float* __restrict__ Wz, const float* __restrict__ Wr,
    const float* __restrict__ W, __bf16* __restrict__ ws)
{
  unsigned gid  = blockIdx.x * blockDim.x + threadIdx.x;
  unsigned frag = gid >> 5;
  int lane = gid & 31;
  int m16  = lane & 15;

  __bf16* w3s  = ws;
  __bf16* xhs  = ws + W3S_ELEMS;
  __bf16* rhxs = xhs + XHS_ELEMS;

  if (frag < 36864u) {
    int w   = frag / (NT_W * KT);
    int rem = frag % (NT_W * KT);
    int nt = rem / KT, kt = rem % KT;
    const float* src = (w == 0) ? Wz : (w == 1) ? Wr : W;
    const float* srow = src + (size_t)(nt * 16 + m16) * K_SZ + kt * 32;
    __bf16* dst = w3s + (size_t)frag * FRAG + lane * 16;
#pragma unroll
    for (int t = 0; t < 16; ++t) dst[t] = (__bf16)srow[bmapK(lane, t)];
  } else if (frag < 61440u) {
    unsigned f = frag - 36864u;
    int mt = f / KT, kt = f % KT;
    int row = mt * 16 + m16;
    __bf16* dst = xhs + (size_t)f * FRAG + lane * 16;
#pragma unroll
    for (int t = 0; t < 16; ++t) {
      int k = kt * 32 + amapK(lane, t);
      float v = (k < IN_SZ) ? x[(size_t)row * IN_SZ + k]
                            : h[(size_t)row * HID_SZ + (k - IN_SZ)];
      dst[t] = (__bf16)v;
    }
  } else if (frag < 69632u) {
    unsigned f = frag - 61440u;
    int mt = f / 32, kt = 64 + (f % 32);     // rhx cols [2048,3072) = x
    int row = mt * 16 + m16;
    __bf16* dst = rhxs + ((size_t)mt * KT + kt) * FRAG + lane * 16;
#pragma unroll
    for (int t = 0; t < 16; ++t) {
      int k = kt * 32 + amapK(lane, t);
      dst[t] = (__bf16)x[(size_t)row * IN_SZ + (k - HID_SZ)];
    }
  }
}

// ---------------- GEMM + fused GRU epilogues --------------------------------
// 256 threads = 8 waves; block tile 128x128; wave tile 64(M) x 32(N).
// Per K-step: block stages 16 fragments (16 KB) into LDS (double-buffered,
// async global->LDS), each wave reads 6 fragments via ds_load and issues
// 8 x v_wmma_f32_16x16x32_bf16.
__global__ __launch_bounds__(256) void gru_gemm(
    const __bf16* __restrict__ As, const __bf16* __restrict__ Bs,
    const float* __restrict__ bias, const float* __restrict__ hprev,
    float* __restrict__ zbuf, __bf16* __restrict__ rhxs,
    float* __restrict__ out0, float* __restrict__ out1, int mode)
{
  __shared__ __bf16 smem[2][16 * FRAG];   // [buf][frag 0..7 = A, 8..15 = B]

  int lane = threadIdx.x & 31;
  int wave = threadIdx.x >> 5;
  int wm = wave & 1;        // 2 waves along M
  int wn = wave >> 1;       // 4 waves along N
  int mtBlock = blockIdx.x * 8;
  int ntBlock = blockIdx.y * 8;
  int mtBase = mtBlock + wm * 4;
  int ntBase = ntBlock + wn * 2;

  // Precompute this thread's 4 copy chunks (16B each, fixed frag/offset).
  const __bf16* gbase[4];
  __bf16* lbase[4];
#pragma unroll
  for (int c = 0; c < 4; ++c) {
    int chunk = threadIdx.x + c * 256;     // 0..1023
    int f     = chunk >> 6;                // fragment 0..15
    int off   = (chunk & 63) * 8;          // element offset within fragment
    if (f < 8) gbase[c] = As + ((size_t)(mtBlock + f) * KT) * FRAG + off;
    else       gbase[c] = Bs + ((size_t)(ntBlock + f - 8) * KT) * FRAG + off;
    lbase[c] = &smem[0][f * FRAG + off];
  }

  auto stage = [&](int buf, int kt) {
#pragma unroll
    for (int c = 0; c < 4; ++c) {
      const __bf16* g = gbase[c] + (size_t)kt * FRAG;
      __bf16* l = lbase[c] + buf * (16 * FRAG);
#if USE_ASYNC_LDS
      __builtin_amdgcn_global_load_async_to_lds_b128((v4i*)g, (v4i*)l, 0, 0);
#else
      *(v4i*)l = *(const v4i*)g;
#endif
    }
  };

  v8f acc[4][2] = {};

  stage(0, 0);
#pragma unroll 2
  for (int kt = 0; kt < KT; ++kt) {
#if USE_ASYNC_LDS
    __builtin_amdgcn_s_wait_asynccnt(0);
#endif
    __syncthreads();
    if (kt + 1 < KT) stage((kt + 1) & 1, kt + 1);   // overlap with compute

    const __bf16* la = &smem[kt & 1][(wm * 4) * FRAG + lane * 16];
    const __bf16* lb = &smem[kt & 1][(8 + wn * 2) * FRAG + lane * 16];
    v16bf a[4], b[2];
    // Issue every fragment load (B first) before any wmma so the waits
    // stagger (dscnt 6/4/2/0) instead of a hard 0 before each pair.
#pragma unroll
    for (int j = 0; j < 2; ++j) b[j] = *(const v16bf*)(lb + j * FRAG);
#pragma unroll
    for (int i = 0; i < 4; ++i) a[i] = *(const v16bf*)(la + i * FRAG);
    sched_fence();
#pragma unroll
    for (int i = 0; i < 4; ++i)
#pragma unroll
      for (int j = 0; j < 2; ++j)
        acc[i][j] = __builtin_amdgcn_wmma_f32_16x16x32_bf16(
            false, a[i], false, b[j], (short)0, acc[i][j], false, false);
  }

  // C layout: lanes0-15 -> M = v, lanes16-31 -> M = v+8; N = lane&15
  int col0 = lane & 15;
  int rhi  = (lane >> 4) * 8;
#pragma unroll
  for (int i = 0; i < 4; ++i) {
#pragma unroll
    for (int j = 0; j < 2; ++j) {
#pragma unroll
      for (int v = 0; v < 8; ++v) {
        int row = (mtBase + i) * 16 + rhi + v;
        int col = (ntBase + j) * 16 + col0;
        size_t idx = (size_t)row * HID_SZ + col;
        float val = acc[i][j][v] + bias[col];
        if (mode == 0) {
          zbuf[idx] = 1.0f / (1.0f + __expf(-val));
        } else if (mode == 1) {
          float r  = 1.0f / (1.0f + __expf(-val));
          float rh = r * hprev[idx];
          // inverse A-map: (row, col) -> (lane', halfword') of swizzled rhx
          int kk = col & 31, ktc = col >> 5;
          int t, hb;
          if (kk < 8)       { t = kk;      hb = 0; }
          else if (kk < 16) { t = kk - 8;  hb = 1; }
          else if (kk < 24) { t = kk - 8;  hb = 0; }
          else              { t = kk - 16; hb = 1; }
          rhxs[((size_t)(row >> 4) * KT + ktc) * FRAG +
               ((row & 15) + 16 * hb) * 16 + t] = (__bf16)rh;
        } else {
          float htl = tanhf(val);
          float z   = zbuf[idx];            // read own element, then overwrite
          float hp  = hprev[idx];
          float ht  = (1.0f - z) * hp + z * htl;
          out0[idx] = ht;
          out1[idx] = ht;
        }
      }
    }
  }
}

extern "C" void kernel_launch(void* const* d_in, const int* in_sizes, int n_in,
                              void* d_out, int out_size, void* d_ws, size_t ws_size,
                              hipStream_t stream)
{
  const float* x  = (const float*)d_in[0];
  const float* h  = (const float*)d_in[1];
  const float* Wz = (const float*)d_in[2];
  const float* bz = (const float*)d_in[3];
  const float* Wr = (const float*)d_in[4];
  const float* br = (const float*)d_in[5];
  const float* W  = (const float*)d_in[6];
  const float* b  = (const float*)d_in[7];
  float*  out = (float*)d_out;
  __bf16* ws  = (__bf16*)d_ws;

  __bf16* w3s  = ws;
  __bf16* xhs  = ws + W3S_ELEMS;
  __bf16* rhxs = xhs + XHS_ELEMS;

  float* zbuf = out;                              // first half of d_out
  float* out1 = out + (size_t)B_SZ * HID_SZ;      // second tuple element

  // 1) convert + swizzle (69632 fragments * 32 lanes)
  gru_prep<<<(69632u * 32u) / 256u, 256, 0, stream>>>(x, h, Wz, Wr, W, ws);

  dim3 grid(B_SZ / 128, HID_SZ / 128);
  size_t wchunk = (size_t)NT_W * KT * FRAG;

  // 2) z gate
  gru_gemm<<<grid, 256, 0, stream>>>(xhs, w3s,            bz, h, zbuf, rhxs, out, out1, 0);
  // 3) r gate -> r*h scattered into swizzled rhx
  gru_gemm<<<grid, 256, 0, stream>>>(xhs, w3s + wchunk,   br, h, zbuf, rhxs, out, out1, 1);
  // 4) candidate + final blend
  gru_gemm<<<grid, 256, 0, stream>>>(rhxs, w3s + 2*wchunk, b, h, zbuf, rhxs, out, out1, 2);
}
